// CommAwareGCN_5858335392390
// MI455X (gfx1250) — compile-verified, matrix operand order checked by reading, not run on hardware
//
#include <hip/hip_runtime.h>
#include <hip/hip_bf16.h>
#include <math.h>

// GCN: collapse per-edge GEMMs to per-node GEMMs + pure scatter-adds.
// GEMMs via v_wmma_f32_16x16x32_f16 (wave32, 16x16 tile, K=32 per op).

typedef __attribute__((ext_vector_type(16))) _Float16 v16h;
typedef __attribute__((ext_vector_type(8)))  float    v8f;

#define IN_CH 64
#define HID   64
#define N_CLS 40

// ---- A fragment: 16x32 f16, lane = {h=lane>>4, m=lane&15}.
// Element j (0..15) holds K = kc*32 + 8*h + j + 8*(j>=8): two contiguous
// 8-float runs -> four float4 loads, converted f32->f16.
__device__ inline v16h load_a_frag(const float* __restrict__ row, int h, int kc) {
  const float* rp = row + kc * 32;
  float4 f0 = *reinterpret_cast<const float4*>(rp + 8 * h);
  float4 f1 = *reinterpret_cast<const float4*>(rp + 8 * h + 4);
  float4 f2 = *reinterpret_cast<const float4*>(rp + 16 + 8 * h);
  float4 f3 = *reinterpret_cast<const float4*>(rp + 16 + 8 * h + 4);
  v16h a;
  a[0]  = (_Float16)f0.x; a[1]  = (_Float16)f0.y; a[2]  = (_Float16)f0.z; a[3]  = (_Float16)f0.w;
  a[4]  = (_Float16)f1.x; a[5]  = (_Float16)f1.y; a[6]  = (_Float16)f1.z; a[7]  = (_Float16)f1.w;
  a[8]  = (_Float16)f2.x; a[9]  = (_Float16)f2.y; a[10] = (_Float16)f2.z; a[11] = (_Float16)f2.w;
  a[12] = (_Float16)f3.x; a[13] = (_Float16)f3.y; a[14] = (_Float16)f3.z; a[15] = (_Float16)f3.w;
  return a;
}

// ---- B fragment: 32x16 f16, lane = column n, same K mapping as A.
// W is row-major [K, wcols]; col >= wcols loads 0 (classifier pad).
__device__ inline v16h load_b_frag(const float* __restrict__ W, int wld, int wcols,
                                   int col, int h, int kc) {
  v16h b;
#pragma unroll
  for (int j = 0; j < 16; ++j) {
    int k = kc * 32 + 8 * h + j + ((j >> 3) << 3);
    float v = (col < wcols) ? W[k * wld + col] : 0.0f;
    b[j] = (_Float16)v;
  }
  return b;
}

// ---- Node-level GEMM: out[N,64] = act(X[N,64] @ W[64,64] + scale*bias)
// One wave per 16-row tile; 2 K-chunks x 4 N-tiles = 8 WMMAs.
__global__ void __launch_bounds__(128)
gemm_node64(const float* __restrict__ X, const float* __restrict__ W,
            const float* __restrict__ bias, const float* __restrict__ degScale,
            float* __restrict__ out, int nrows, int do_relu) {
  int wave = blockIdx.x * (blockDim.x >> 5) + (threadIdx.x >> 5);
  int row0 = wave * 16;
  if (row0 >= nrows) return;                    // wave-uniform: EXEC stays all-1s
  int lane = threadIdx.x & 31;
  int n = lane & 15, h = lane >> 4;

  float s[8];
#pragma unroll
  for (int r = 0; r < 8; ++r)
    s[r] = degScale ? degScale[row0 + h * 8 + r] : 1.0f;

  v8f acc[4];
#pragma unroll
  for (int nt = 0; nt < 4; ++nt) {
    float bcol = bias[nt * 16 + n];
#pragma unroll
    for (int r = 0; r < 8; ++r) acc[nt][r] = s[r] * bcol;
  }

  const float* rowA = X + (size_t)(row0 + n) * 64;
#pragma unroll
  for (int kc = 0; kc < 2; ++kc) {
    v16h a = load_a_frag(rowA, h, kc);
#pragma unroll
    for (int nt = 0; nt < 4; ++nt) {
      v16h b = load_b_frag(W, 64, 64, nt * 16 + n, h, kc);
      acc[nt] = __builtin_amdgcn_wmma_f32_16x16x32_f16(
          false, a, false, b, (short)0, acc[nt], false, false);
    }
  }

#pragma unroll
  for (int nt = 0; nt < 4; ++nt)
#pragma unroll
    for (int r = 0; r < 8; ++r) {
      float v = acc[nt][r];
      if (do_relu) v = fmaxf(v, 0.0f);
      int row = row0 + h * 8 + r;
      out[(size_t)row * 64 + nt * 16 + n] = v;
    }
}

// ---- Classifier + softmax: out[N,40] = softmax(X[N,64] @ Wf[64,40] + bf)
// 3 N-tiles (48 cols, cols>=40 masked via -1e30 bias / zero weights).
__global__ void __launch_bounds__(128)
classify_softmax(const float* __restrict__ X, const float* __restrict__ Wf,
                 const float* __restrict__ bf, float* __restrict__ out, int nrows) {
  int wave = blockIdx.x * (blockDim.x >> 5) + (threadIdx.x >> 5);
  int row0 = wave * 16;
  if (row0 >= nrows) return;
  int lane = threadIdx.x & 31;
  int n = lane & 15, h = lane >> 4;

  v8f acc[3];
#pragma unroll
  for (int nt = 0; nt < 3; ++nt) {
    int col = nt * 16 + n;
    float bcol = (col < N_CLS) ? bf[col] : -1e30f;
#pragma unroll
    for (int r = 0; r < 8; ++r) acc[nt][r] = bcol;
  }

  const float* rowA = X + (size_t)(row0 + n) * 64;
#pragma unroll
  for (int kc = 0; kc < 2; ++kc) {
    v16h a = load_a_frag(rowA, h, kc);
#pragma unroll
    for (int nt = 0; nt < 3; ++nt) {
      v16h b = load_b_frag(Wf, N_CLS, N_CLS, nt * 16 + n, h, kc);
      acc[nt] = __builtin_amdgcn_wmma_f32_16x16x32_f16(
          false, a, false, b, (short)0, acc[nt], false, false);
    }
  }

#pragma unroll
  for (int r = 0; r < 8; ++r) {
    float v0 = acc[0][r], v1 = acc[1][r], v2 = acc[2][r];
    float m = fmaxf(v0, fmaxf(v1, v2));
#pragma unroll
    for (int mask = 1; mask < 16; mask <<= 1)
      m = fmaxf(m, __shfl_xor(m, mask, 16));     // reduce across 16-lane half
    float e0 = __expf(v0 - m), e1 = __expf(v1 - m), e2 = __expf(v2 - m);
    float sum = e0 + e1 + e2;                    // masked cols contribute exp(-1e30)=0
#pragma unroll
    for (int mask = 1; mask < 16; mask <<= 1)
      sum += __shfl_xor(sum, mask, 16);
    float inv = 1.0f / sum;
    int row = row0 + h * 8 + r;
    out[(size_t)row * N_CLS + n]      = e0 * inv;
    out[(size_t)row * N_CLS + 16 + n] = e1 * inv;
    if (n < 8) out[(size_t)row * N_CLS + 32 + n] = e2 * inv;
  }
}

// ---- In-degree (as float, so layer-2 bias scale needs no convert).
__global__ void deg_count(const int* __restrict__ ei, int E, float* __restrict__ deg) {
  int t = blockIdx.x * blockDim.x + threadIdx.x;
  if (t < E) atomicAdd(&deg[ei[E + t]], 1.0f);
}

// ---- Scatter-add: acc[dst] += feat[src], 64 ch/edge, 64 threads/edge.
// Gather read = coalesced 256B; atomics = 64 consecutive lanes into L2.
__global__ void scatter64(const float* __restrict__ feat, const int* __restrict__ ei,
                          int E, float* __restrict__ acc) {
  long long t = (long long)blockIdx.x * blockDim.x + threadIdx.x;
  int e = (int)(t >> 6);
  int c = (int)(t & 63);
  if (e < E) {
    int s = ei[e];
    int d = ei[E + e];
    atomicAdd(&acc[(size_t)d * 64 + c], feat[(size_t)s * 64 + c]);
  }
}

extern "C" void kernel_launch(void* const* d_in, const int* in_sizes, int n_in,
                              void* d_out, int out_size, void* d_ws, size_t ws_size,
                              hipStream_t stream) {
  const float* X  = (const float*)d_in[0];
  const int*   ei = (const int*)d_in[1];     // [2,E]: row 0 = src, row 1 = dst
  const float* W1 = (const float*)d_in[2];
  const float* b1 = (const float*)d_in[3];
  const float* W2 = (const float*)d_in[4];
  const float* b2 = (const float*)d_in[5];
  const float* Wf = (const float*)d_in[6];
  const float* bf = (const float*)d_in[7];
  float* out = (float*)d_out;

  int N = in_sizes[0] / IN_CH;
  int E = in_sizes[1] / 2;

  // Workspace: H (reused for H1 then H2), A1, A2, deg.  3*25.6MB + 0.4MB.
  float* H   = (float*)d_ws;
  float* A1  = H  + (size_t)N * HID;
  float* A2  = A1 + (size_t)N * HID;
  float* deg = A2 + (size_t)N * HID;

  hipMemsetAsync(A1,  0, (size_t)N * HID * sizeof(float), stream);
  hipMemsetAsync(A2,  0, (size_t)N * HID * sizeof(float), stream);
  hipMemsetAsync(deg, 0, (size_t)N * sizeof(float), stream);

  int tiles = (N + 15) / 16;
  int gblocks = (tiles + 3) / 4;               // 4 waves (128 thr) per block

  // H1 = relu(X @ W1 + b1)   [node-level, not edge-level]
  gemm_node64<<<gblocks, 128, 0, stream>>>(X, W1, b1, nullptr, H, N, 1);

  // deg[n] = |{e : dst[e]==n}|
  deg_count<<<(E + 255) / 256, 256, 0, stream>>>(ei, E, deg);

  // A1[dst] += H1[src]
  long long st = (long long)E * 64;
  int sblocks = (int)((st + 255) / 256);
  scatter64<<<sblocks, 256, 0, stream>>>(H, ei, E, A1);

  // A2[dst] += A1[src]
  scatter64<<<sblocks, 256, 0, stream>>>(A1, ei, E, A2);

  // H2 = A2 @ W2 + deg*b2
  gemm_node64<<<gblocks, 128, 0, stream>>>(A2, W2, b2, deg, H, N, 0);

  // out = softmax(H2 @ Wf + bf)
  classify_softmax<<<gblocks, 128, 0, stream>>>(H, Wf, bf, out, N);
}
